// Model_29944511987744
// MI455X (gfx1250) — compile-verified, matrix operand order checked by reading, not run on hardware
//
#include <hip/hip_runtime.h>

// ---------------------------------------------------------------------------
// Problem: two MLP softmax branches + diagonal max-min aggregation.
//   B=256, IN=1024, S=512.  out[b,k] = max_{i-j=k-(S-1)} min(a[b,i], b[b,j])
// Strategy for MI455X (gfx1250, wave32, WMMA):
//   * stage x / W in bf16 (W transposed to [N,K]) so WMMA frags are b128 loads
//   * v_wmma_f32_16x16x32_bf16 for both GEMMs, f32 accum, fused bias/relu
//   * software-pipelined K loop (double-buffered fragments) so fragment loads
//     for chunk n+1 are in flight while chunk n's 4 WMMAs issue back-to-back
//   * VALU kernels for softmax + diagonal min/max (tiny: ~67M ops)
// ---------------------------------------------------------------------------

typedef __attribute__((ext_vector_type(16))) __bf16        v16bf;
typedef __attribute__((ext_vector_type(8)))  float         v8f;
typedef __attribute__((ext_vector_type(4)))  unsigned int  u32x4;

union Frag {
    v16bf v;
    u32x4 q[2];
};

static __device__ inline void load_frag(Frag& f, const unsigned short* p) {
    f.q[0] = *(const u32x4*)(p);
    f.q[1] = *(const u32x4*)(p + 16);
}

static __device__ inline unsigned short f32_to_bf16_rne(float f) {
    union { float f; unsigned int u; } c;
    c.f = f;
    unsigned int u = c.u;
    u += 0x7FFFu + ((u >> 16) & 1u);        // round-to-nearest-even
    return (unsigned short)(u >> 16);
}

// ---------------------------------------------------------------------------
// 1) elementwise f32 -> bf16 (for x)
// ---------------------------------------------------------------------------
__global__ __launch_bounds__(256)
void cvt_f32_bf16_kernel(const float* __restrict__ in,
                         unsigned short* __restrict__ out, int n) {
    int i = blockIdx.x * blockDim.x + threadIdx.x;
    if (i < n) out[i] = f32_to_bf16_rne(in[i]);
}

// ---------------------------------------------------------------------------
// 2) LDS-tiled transpose + convert: W[K,N] f32  ->  WT[N,K] bf16
//    (coalesced on both the read and the write side)
// ---------------------------------------------------------------------------
__global__ __launch_bounds__(256)
void transpose_f32_to_bf16(const float* __restrict__ W,
                           unsigned short* __restrict__ WT, int K, int N) {
    __shared__ float tile[32][33];
    const int n0 = blockIdx.x * 32;
    const int k0 = blockIdx.y * 32;
    for (int r = threadIdx.y; r < 32; r += 8)
        tile[r][threadIdx.x] = W[(size_t)(k0 + r) * N + (n0 + threadIdx.x)];
    __syncthreads();
    for (int r = threadIdx.y; r < 32; r += 8)
        WT[(size_t)(n0 + r) * K + (k0 + threadIdx.x)] =
            f32_to_bf16_rne(tile[threadIdx.x][r]);
}

// ---------------------------------------------------------------------------
// 3) bf16 WMMA GEMM:  O = act(A[ M x K ] * BT[ N x K ]^T + bias)
//    Block: 256 threads = 8 waves in a 4(M) x 2(N) grid.
//    Block tile: 64(M) x 128(N). Wave tile: 16(M) x 64(N) = 4 accumulators,
//    sharing one A fragment per K-step. Both branches via blockIdx.z.
//
//    bf16 A-frag layout per ISA 7.12.2 (lane half h = lane>>4, m = lane&15):
//      elements 0..7  : K = k0 + 8h .. k0 + 8h + 7        (one b128 load)
//      elements 8..15 : K = k0 + 16 + 8h .. k0 + 16+8h+7  (one b128 load)
//    B frag mirrors it on column N = tileN + 16t + (lane&15), contiguous in K
//    because W was pre-transposed to [N,K].
//    C/D layout: element r -> M = 8h + r, N = lane&15.
//
//    K loop is double-buffered: fragments for chunk n+1 are loaded before the
//    WMMAs of chunk n, keeping ~10 b128 loads in flight behind the matrix ops.
//    Requires K % 64 == 0 (K = 1024 here).
// ---------------------------------------------------------------------------
static __device__ inline void store_out(unsigned short* O, size_t idx, float v) {
    O[idx] = f32_to_bf16_rne(v);
}
static __device__ inline void store_out(float* O, size_t idx, float v) {
    O[idx] = v;
}

template <bool RELU, typename OutT>
__global__ __launch_bounds__(256)
void gemm_bf16_wmma(const unsigned short* __restrict__ Aa,
                    const unsigned short* __restrict__ Ab,
                    const unsigned short* __restrict__ BTa,
                    const unsigned short* __restrict__ BTb,
                    const float* __restrict__ biasAp,
                    const float* __restrict__ biasBp,
                    OutT* __restrict__ Oa,
                    OutT* __restrict__ Ob,
                    int N, int K) {
    const unsigned short* __restrict__ A    = blockIdx.z ? Ab     : Aa;
    const unsigned short* __restrict__ BT   = blockIdx.z ? BTb    : BTa;
    const float*          __restrict__ bias = blockIdx.z ? biasBp : biasAp;
    OutT*                 __restrict__ O    = blockIdx.z ? Ob     : Oa;

    const int lane = threadIdx.x & 31;
    const int wave = threadIdx.x >> 5;    // 0..7
    const int wm   = wave >> 1;           // 0..3
    const int wn   = wave & 1;            // 0..1
    const int half = lane >> 4;           // 0/1
    const int l16  = lane & 15;

    const int tileM = blockIdx.y * 64 + wm * 16;
    const int tileN = blockIdx.x * 128 + wn * 64;

    // lane-private base pointers (already include the half*8 K-phase)
    const unsigned short* ap = A + (size_t)(tileM + l16) * K + half * 8;
    const unsigned short* bp0 = BT + (size_t)(tileN + l16) * K + half * 8;
    const size_t colStride = (size_t)16 * K;   // next 16-column group

    v8f acc[4] = {};

    Frag a0, a1;
    Frag b0[4], b1[4];

    // prologue: chunk at k=0
    load_frag(a0, ap);
#pragma unroll
    for (int t = 0; t < 4; ++t) load_frag(b0[t], bp0 + colStride * t);

    int k0 = 0;
    for (; k0 < K - 64; k0 += 64) {
        // load chunk k0+32 while chunk k0 computes
        load_frag(a1, ap + k0 + 32);
        __builtin_prefetch(ap + k0 + 96, 0, 1);   // -> global_prefetch_b8
#pragma unroll
        for (int t = 0; t < 4; ++t) load_frag(b1[t], bp0 + colStride * t + k0 + 32);
#pragma unroll
        for (int t = 0; t < 4; ++t)
            acc[t] = __builtin_amdgcn_wmma_f32_16x16x32_bf16(
                false, a0.v, false, b0[t].v, (short)0, acc[t], false, false);

        // load chunk k0+64 while chunk k0+32 computes
        load_frag(a0, ap + k0 + 64);
#pragma unroll
        for (int t = 0; t < 4; ++t) load_frag(b0[t], bp0 + colStride * t + k0 + 64);
#pragma unroll
        for (int t = 0; t < 4; ++t)
            acc[t] = __builtin_amdgcn_wmma_f32_16x16x32_bf16(
                false, a1.v, false, b1[t].v, (short)0, acc[t], false, false);
    }
    // epilogue: final 64 columns of K (chunks k0 and k0+32)
    load_frag(a1, ap + k0 + 32);
#pragma unroll
    for (int t = 0; t < 4; ++t) load_frag(b1[t], bp0 + colStride * t + k0 + 32);
#pragma unroll
    for (int t = 0; t < 4; ++t)
        acc[t] = __builtin_amdgcn_wmma_f32_16x16x32_bf16(
            false, a0.v, false, b0[t].v, (short)0, acc[t], false, false);
#pragma unroll
    for (int t = 0; t < 4; ++t)
        acc[t] = __builtin_amdgcn_wmma_f32_16x16x32_bf16(
            false, a1.v, false, b1[t].v, (short)0, acc[t], false, false);

    // Epilogue: bias (+ReLU), store per documented C layout
    const int mBase = tileM + half * 8;
#pragma unroll
    for (int t = 0; t < 4; ++t) {
        const int n  = tileN + t * 16 + l16;
        const float bv = bias[n];
#pragma unroll
        for (int r = 0; r < 8; ++r) {
            float v = acc[t][r] + bv;
            if (RELU) v = fmaxf(v, 0.0f);
            store_out(O, (size_t)(mBase + r) * N + n, v);
        }
    }
}

// ---------------------------------------------------------------------------
// 4) in-place row softmax over 512 columns; grid.x = row, grid.y = branch
// ---------------------------------------------------------------------------
__global__ __launch_bounds__(256)
void softmax512_kernel(float* la, float* lb) {
    float* p = (blockIdx.y ? lb : la) + (size_t)blockIdx.x * 512;
    const int t = threadIdx.x;

    float v0 = p[t], v1 = p[t + 256];
    __shared__ float redm[8], reds[8];

    // block max
    float m = fmaxf(v0, v1);
#pragma unroll
    for (int o = 16; o; o >>= 1) m = fmaxf(m, __shfl_xor(m, o, 32));
    if ((t & 31) == 0) redm[t >> 5] = m;
    __syncthreads();
    if (t == 0) {
        float w = redm[0];
        for (int i = 1; i < 8; ++i) w = fmaxf(w, redm[i]);
        redm[0] = w;
    }
    __syncthreads();
    m = redm[0];

    // exp + block sum
    float e0 = __expf(v0 - m);
    float e1 = __expf(v1 - m);
    float s = e0 + e1;
#pragma unroll
    for (int o = 16; o; o >>= 1) s += __shfl_xor(s, o, 32);
    if ((t & 31) == 0) reds[t >> 5] = s;
    __syncthreads();
    if (t == 0) {
        float w = 0.0f;
        for (int i = 0; i < 8; ++i) w += reds[i];
        reds[0] = w;
    }
    __syncthreads();
    const float inv = 1.0f / reds[0];

    p[t]       = e0 * inv;
    p[t + 256] = e1 * inv;
}

// ---------------------------------------------------------------------------
// 5) diagonal max-of-min: out[b,k] = max_{i-j=k-511} min(a[b,i], b[b,j])
//    one block per batch row; a/b rows staged in LDS
// ---------------------------------------------------------------------------
__global__ __launch_bounds__(256)
void diag_maxmin_kernel(const float* __restrict__ la,
                        const float* __restrict__ lb,
                        float* __restrict__ out) {
    __shared__ float sa[512], sb[512];
    const float* a = la + (size_t)blockIdx.x * 512;
    const float* b = lb + (size_t)blockIdx.x * 512;
    const int t = threadIdx.x;
    sa[t] = a[t];  sa[t + 256] = a[t + 256];
    sb[t] = b[t];  sb[t + 256] = b[t + 256];
    __syncthreads();

    for (int k = t; k < 1023; k += 256) {
        const int d   = k - 511;                 // i - j
        const int ilo = d > 0 ? d : 0;
        const int ihi = d > 0 ? 511 : 511 + d;   // inclusive
        float m = -3.402823466e+38f;
        for (int i = ilo; i <= ihi; ++i) {
            float v = fminf(sa[i], sb[i - d]);
            m = fmaxf(m, v);
        }
        out[(size_t)blockIdx.x * 1023 + k] = m;
    }
}

// ---------------------------------------------------------------------------
// host launcher
// ---------------------------------------------------------------------------
extern "C" void kernel_launch(void* const* d_in, const int* in_sizes, int n_in,
                              void* d_out, int out_size, void* d_ws, size_t ws_size,
                              hipStream_t stream) {
    (void)in_sizes; (void)n_in; (void)out_size; (void)ws_size;

    const int B = 256, IN = 1024, S = 512;

    const float* x   = (const float*)d_in[0];
    const float* W1a = (const float*)d_in[1];
    const float* b1a = (const float*)d_in[2];
    const float* W2a = (const float*)d_in[3];
    const float* b2a = (const float*)d_in[4];
    const float* W1b = (const float*)d_in[5];
    const float* b1b = (const float*)d_in[6];
    const float* W2b = (const float*)d_in[7];
    const float* b2b = (const float*)d_in[8];
    float* out = (float*)d_out;

    // workspace carve-out (~8.5 MB, 256B aligned slices)
    size_t off = 0;
    char* base = (char*)d_ws;
    auto carve = [&](size_t bytes) {
        char* p = base + off;
        off += (bytes + 255) & ~(size_t)255;
        return p;
    };
    unsigned short* xb    = (unsigned short*)carve((size_t)B * IN * 2);
    unsigned short* w1aT  = (unsigned short*)carve((size_t)IN * IN * 2);
    unsigned short* w1bT  = (unsigned short*)carve((size_t)IN * IN * 2);
    unsigned short* w2aT  = (unsigned short*)carve((size_t)S * IN * 2);
    unsigned short* w2bT  = (unsigned short*)carve((size_t)S * IN * 2);
    unsigned short* ha    = (unsigned short*)carve((size_t)B * IN * 2);
    unsigned short* hb    = (unsigned short*)carve((size_t)B * IN * 2);
    float*          la    = (float*)carve((size_t)B * S * 4);
    float*          lb    = (float*)carve((size_t)B * S * 4);

    // 1) stage x to bf16
    {
        int n = B * IN;
        cvt_f32_bf16_kernel<<<dim3((n + 255) / 256), dim3(256), 0, stream>>>(x, xb, n);
    }
    // 2) transpose+convert weights: W[K,N] -> WT[N,K] bf16
    {
        dim3 blk(32, 8);
        transpose_f32_to_bf16<<<dim3(IN / 32, IN / 32), blk, 0, stream>>>(W1a, w1aT, IN, IN);
        transpose_f32_to_bf16<<<dim3(IN / 32, IN / 32), blk, 0, stream>>>(W1b, w1bT, IN, IN);
        transpose_f32_to_bf16<<<dim3(S  / 32, IN / 32), blk, 0, stream>>>(W2a, w2aT, IN, S);
        transpose_f32_to_bf16<<<dim3(S  / 32, IN / 32), blk, 0, stream>>>(W2b, w2bT, IN, S);
    }
    // 3) h = relu(x @ W1 + b1), bf16 out, both branches (grid.z)
    gemm_bf16_wmma<true, unsigned short>
        <<<dim3(IN / 128, B / 64, 2), dim3(256), 0, stream>>>(
            xb, xb, w1aT, w1bT, b1a, b1b, ha, hb, IN, IN);
    // 4) logits = h @ W2 + b2, f32 out, both branches
    gemm_bf16_wmma<false, float>
        <<<dim3(S / 128, B / 64, 2), dim3(256), 0, stream>>>(
            ha, hb, w2aT, w2bT, b2a, b2b, la, lb, S, IN);
    // 5) softmax rows in place
    softmax512_kernel<<<dim3(B, 2), dim3(256), 0, stream>>>(la, lb);
    // 6) diagonal max-of-min
    diag_maxmin_kernel<<<dim3(B), dim3(256), 0, stream>>>(la, lb, out);
}
